// GatedDeltaNetAttention_65180423684453
// MI455X (gfx1250) — matrix operand order repaired and checked by reading, not compiled, not run on hardware
//
#include <hip/hip_runtime.h>
#include <cstdint>
#include <cstddef>

// ---------------------------------------------------------------------------
// Gated DeltaNet attention for MI455X (gfx1250), bf16 WMMA path +
// async global->LDS staging (double-buffered GEMM).
// B=2, T=2048, DIM=2048, H=16, Dk=Dv=128, K(conv)=4, chunk=64.
// ---------------------------------------------------------------------------

#define BB   2
#define TT   2048
#define DIMM 2048
#define HH   16
#define DK   128
#define DV   128
#define QD   2048   // H*Dk
#define VD   2048   // H*Dv
#define KCONV 4
#define CHUNK 64

typedef __attribute__((ext_vector_type(16))) __bf16 v16bf;
typedef __attribute__((ext_vector_type(8)))  float  v8f;

// ----------------------- CDNA5 async global->LDS helpers -------------------
// GLOBAL_LOAD_ASYNC_TO_LDS_B128 (GV mode): per-lane 16B copy global -> LDS,
// tracked by ASYNCcnt (CDNA5 ISA).
__device__ __forceinline__ uint32_t lds_off(const void* p) {
  return (uint32_t)(uintptr_t)(__attribute__((address_space(3))) const void*)p;
}
__device__ __forceinline__ void async_g2l_b128(uint32_t loff, const void* g) {
  asm volatile("global_load_async_to_lds_b128 %0, %1, off"
               :: "v"(loff), "v"(g) : "memory");
}
__device__ __forceinline__ void wait_async0() {
  asm volatile("s_wait_asynccnt 0x0" ::: "memory");
}

// ---------------- WMMA fragment helpers (wave32, 16x16x32 bf16) ------------
// A matrix 16x32 bf16, row-major source with 'stride' elements per row.
//   hi=0: e0..7 -> K=e,     e8..15 -> K=e+8   (i.e. 0..7,16..23)
//   hi=1: e0..7 -> K=e+8,   e8..15 -> K=e+16  (i.e. 8..15,24..31)
__device__ __forceinline__ v16bf frag_a(const __bf16* base, int stride, int lane) {
  int m = lane & 15, hi = lane >> 4;
  const __bf16* row = base + (size_t)m * stride;
  v16bf f;
#pragma unroll
  for (int e = 0; e < 16; ++e) {
    int k = e + 8 * hi + ((e & 8) ? 8 : 0);
    f[e] = row[k];
  }
  return f;
}

// Same, but scale the whole row (one m per lane) by rowscale in f32.
__device__ __forceinline__ v16bf frag_a_scaled(const __bf16* base, int stride,
                                               int lane, float rowscale) {
  int m = lane & 15, hi = lane >> 4;
  const __bf16* row = base + (size_t)m * stride;
  v16bf f;
#pragma unroll
  for (int e = 0; e < 16; ++e) {
    int k = e + 8 * hi + ((e & 8) ? 8 : 0);
    f[e] = (__bf16)((float)row[k] * rowscale);
  }
  return f;
}

// B matrix 32x16 bf16, staged TRANSPOSED in LDS as Bt[n][k] (stride elems/row).
__device__ __forceinline__ v16bf frag_bt(const __bf16* base, int stride, int lane) {
  int n = lane & 15, hi = lane >> 4;
  const __bf16* row = base + (size_t)n * stride;
  v16bf f;
#pragma unroll
  for (int e = 0; e < 16; ++e) f[e] = row[e + 8 * hi + ((e & 8) ? 8 : 0)];
  return f;
}

__device__ __forceinline__ v8f wmma_bf16(v16bf a, v16bf b, v8f c) {
  return __builtin_amdgcn_wmma_f32_16x16x32_bf16(false, a, false, b,
                                                 (short)0, c, false, false);
}

__device__ __forceinline__ float sigmoidf_(float x) {
  return 1.0f / (1.0f + __expf(-x));
}

// ---------------------------- f32 -> bf16 convert --------------------------
__global__ void cvt_f32_bf16(const float* __restrict__ in,
                             __bf16* __restrict__ out, size_t n) {
  size_t i = (size_t)blockIdx.x * blockDim.x + threadIdx.x;
  if (i < n) out[i] = (__bf16)in[i];
}

// ------------------------------ GEMM (bf16 WMMA) ---------------------------
// C[M,N] f32 = A[M,K] bf16 @ B[K,N] bf16.  Tile 128x128, 256 threads / 8 waves,
// each wave one 16-row strip x 128 cols.  A tile staged by async global->LDS
// b128 copies, B tile transposed through VGPRs; double-buffered K pipeline.
__global__ __launch_bounds__(256)
void gemm_bf16_wmma(const __bf16* __restrict__ A, const __bf16* __restrict__ Bm,
                    float* __restrict__ C, int M, int N, int Kd) {
  __shared__ __bf16 As[2][128][32];   // [m][k]
  __shared__ __bf16 Bs[2][128][32];   // transposed: [n][k]
  const int tid  = threadIdx.x;
  const int wave = tid >> 5, lane = tid & 31;
  const int bm = blockIdx.y * 128, bn = blockIdx.x * 128;

  v8f acc[8];
#pragma unroll
  for (int i = 0; i < 8; ++i)
#pragma unroll
    for (int e = 0; e < 8; ++e) acc[i][e] = 0.0f;

  // A tile stage: 128 rows x 32 bf16 (64B) = 512 x 16B chunks, 2 per thread.
  auto stageA = [&](int p, int k0) {
#pragma unroll
    for (int c = tid; c < 512; c += 256) {
      int r = c >> 2, seg = (c & 3) * 8;
      async_g2l_b128(lds_off(&As[p][r][seg]),
                     A + (size_t)(bm + r) * Kd + k0 + seg);
    }
  };
  // B tile stage (transpose): coalesced global row read, LDS scatter.
  auto stageB = [&](int p, int k0) {
    int kk = tid >> 3, nb = (tid & 7) * 16;
    const __bf16* g = Bm + (size_t)(k0 + kk) * N + bn + nb;
#pragma unroll
    for (int e = 0; e < 16; ++e) Bs[p][nb + e][kk] = g[e];
  };

  stageA(0, 0);
  stageB(0, 0);
  wait_async0();
  __syncthreads();

  int p = 0;
  for (int k0 = 0; k0 < Kd; k0 += 32, p ^= 1) {
    if (k0 + 32 < Kd) {        // prefetch next K-slice into other buffer
      stageA(p ^ 1, k0 + 32);
      stageB(p ^ 1, k0 + 32);
    }
    v16bf af = frag_a(&As[p][wave * 16][0], 32, lane);
#pragma unroll
    for (int nt = 0; nt < 8; ++nt) {
      v16bf bf_ = frag_bt(&Bs[p][nt * 16][0], 32, lane);
      acc[nt] = wmma_bf16(af, bf_, acc[nt]);
    }
    wait_async0();             // next A slice resident
    __syncthreads();           // B stores done; everyone finished reading cur
  }

  const int n = lane & 15, hi = lane >> 4;
#pragma unroll
  for (int nt = 0; nt < 8; ++nt)
#pragma unroll
    for (int r = 0; r < 8; ++r) {
      int m = bm + wave * 16 + r + 8 * hi;
      C[(size_t)m * N + bn + nt * 16 + n] = acc[nt][r];
    }
}

// ------------------- causal depthwise conv (K=4) + bias + act --------------
__global__ void conv_causal(const float* __restrict__ pre,
                            const float* __restrict__ w,
                            const float* __restrict__ bias,
                            __bf16* __restrict__ out, int mode) {
  size_t i = (size_t)blockIdx.x * blockDim.x + threadIdx.x;  // over B*T*C
  const int C = DIMM;
  int c = (int)(i % C);
  int t = (int)((i / C) % TT);
  size_t b = i / ((size_t)C * TT);
  float acc = bias[c];
#pragma unroll
  for (int j = 0; j < KCONV; ++j) {
    int ts = t - (KCONV - 1) + j;
    if (ts >= 0) acc += w[c * KCONV + j] * pre[(b * TT + ts) * C + c];
  }
  if (mode) acc = acc * sigmoidf_(acc);  // SiLU
  out[i] = (__bf16)acc;
}

// -------------------- per-head L2 normalize of k (wave32) ------------------
__global__ __launch_bounds__(256)
void knorm(__bf16* __restrict__ k) {
  int wave = threadIdx.x >> 5, lane = threadIdx.x & 31;
  int wg = blockIdx.x * 8 + wave;          // over B*T*H
  int h = wg % HH;
  size_t bt = (size_t)(wg / HH);
  __bf16* p = k + bt * QD + h * DK;
  float v0[4], ss = 0.0f;
#pragma unroll
  for (int e = 0; e < 4; ++e) {
    v0[e] = (float)p[lane * 4 + e];
    ss += v0[e] * v0[e];
  }
#pragma unroll
  for (int off = 16; off > 0; off >>= 1) ss += __shfl_xor(ss, off, 32);
  float inv = 1.0f / fmaxf(__fsqrt_rn(ss), 1e-12f);
#pragma unroll
  for (int e = 0; e < 4; ++e) p[lane * 4 + e] = (__bf16)(v0[e] * inv);
}

// ------------------------ chunked gated linear attention -------------------
// One block per (b,h); 8 waves; sequential over T/CHUNK chunks.
// State S[v][d] (f32, LDS).  Per chunk:
//   scores  A[t][s] = beta * lam^(t-s) * (q_t . k_s), s<=t   (WMMA + mask)
//   out     O[t][v] = (A @ V)[t][v] + lam^(t+1) * (Q @ S_bf^T)[t][v]
//   carry   S = lam^C * S + (beta*lam^(C-1-s) K)^T V          (WMMA, C-acc)
__global__ __launch_bounds__(256)
void chunk_rec(const __bf16* __restrict__ q, const __bf16* __restrict__ k,
               const __bf16* __restrict__ v, const float* __restrict__ beta,
               const float* __restrict__ A_log, float* __restrict__ out) {
  extern __shared__ char smem_raw[];
  float*  Sf  = (float*)smem_raw;            // [128][128]  (v,d)  64KB
  __bf16* Sbf = (__bf16*)(Sf + DV * DK);     // [128][128]  (v,d)  32KB
  __bf16* Qs  = Sbf + DV * DK;               // [64][128]   (t,d)  16KB
  __bf16* Ks  = Qs + CHUNK * DK;             // [64][128]   (s,d)  16KB
  __bf16* Kt  = Ks + CHUNK * DK;             // [128][64]   (d,s)  16KB  scaled
  __bf16* Vs  = Kt + DK * CHUNK;             // [128][64]   (v,s)  16KB
  __bf16* Abf = Vs + DV * CHUNK;             // [64][64]    (t,s)   8KB

  const int bh = blockIdx.x;
  const int b = bh / HH, h = bh % HH;
  const float lam  = sigmoidf_(A_log[h]);
  const float bet  = sigmoidf_(beta[h]);
  const float llam = __logf(lam);
  const float lamC = __expf(llam * (float)CHUNK);

  const int tid = threadIdx.x;
  const int wave = tid >> 5, lane = tid & 31;
  const int n_ = lane & 15, hi_ = lane >> 4;

  for (int i = tid; i < DV * DK; i += 256) Sf[i] = 0.0f;
  __syncthreads();

  for (int c0 = 0; c0 < TT; c0 += CHUNK) {
    // ---- stage Q,K via async global->LDS (row-major, no transform) ----
    const __bf16* qg = q + ((size_t)(b * TT + c0)) * QD + h * DK;
    const __bf16* kg = k + ((size_t)(b * TT + c0)) * QD + h * DK;
#pragma unroll
    for (int c = tid; c < 1024; c += 256) {       // 64 rows x 16 chunks of 16B
      int t = c >> 4, seg = (c & 15) * 8;
      async_g2l_b128(lds_off(Qs + t * DK + seg), qg + (size_t)t * QD + seg);
      async_g2l_b128(lds_off(Ks + t * DK + seg), kg + (size_t)t * QD + seg);
    }
    // ---- V transpose staging through VGPRs ----
    for (int i = tid; i < CHUNK * DV; i += 256) {
      int t = i >> 7, d = i & 127;
      Vs[d * CHUNK + t] = v[((size_t)(b * TT + c0 + t)) * VD + h * DV + d];
    }
    // ---- state snapshot (pre-update) to bf16 ----
    for (int i = tid; i < DV * DK; i += 256) Sbf[i] = (__bf16)Sf[i];
    wait_async0();
    __syncthreads();
    // ---- Kt[d][s] = beta * lam^(C-1-s) * K[s][d] (from LDS Ks) ----
    for (int i = tid; i < CHUNK * DK; i += 256) {
      int t = i >> 7, d = i & 127;
      Kt[d * CHUNK + t] =
          (__bf16)((float)Ks[i] * bet * __expf(llam * (float)(CHUNK - 1 - t)));
    }
    __syncthreads();

    // ---- scores: 64x64, K=128; mask+scale -> Abf ----
    for (int tt = wave; tt < 16; tt += 8) {
      int tm = (tt >> 2) * 16;   // t block
      int tn = (tt & 3) * 16;    // s block
      v8f acc;
#pragma unroll
      for (int e = 0; e < 8; ++e) acc[e] = 0.0f;
#pragma unroll
      for (int kk = 0; kk < DK; kk += 32) {
        v16bf af  = frag_a(Qs + tm * DK + kk, DK, lane);
        v16bf bf_ = frag_bt(Ks + tn * DK + kk, DK, lane);
        acc = wmma_bf16(af, bf_, acc);
      }
#pragma unroll
      for (int r = 0; r < 8; ++r) {
        int t = tm + r + 8 * hi_, s = tn + n_;
        float val = (t >= s) ? acc[r] * bet * __expf(llam * (float)(t - s)) : 0.0f;
        Abf[t * CHUNK + s] = (__bf16)val;
      }
    }
    __syncthreads();

    // ---- output: O = A@V + lam^(t+1) Q @ Sbf^T ----
    for (int tt = wave; tt < 32; tt += 8) {
      int tm = (tt >> 3) * 16;   // t block
      int tn = (tt & 7) * 16;    // v block
      v8f acc;
#pragma unroll
      for (int e = 0; e < 8; ++e) acc[e] = 0.0f;
#pragma unroll
      for (int kk = 0; kk < CHUNK; kk += 32) {          // intra
        v16bf af  = frag_a(Abf + tm * CHUNK + kk, CHUNK, lane);
        v16bf bf_ = frag_bt(Vs + tn * CHUNK + kk, CHUNK, lane);
        acc = wmma_bf16(af, bf_, acc);
      }
      float rs = __expf(llam * (float)(tm + (lane & 15) + 1));
#pragma unroll
      for (int kk = 0; kk < DK; kk += 32) {             // inter (carried state)
        v16bf af  = frag_a_scaled(Qs + tm * DK + kk, DK, lane, rs);
        v16bf bf_ = frag_bt(Sbf + tn * DK + kk, DK, lane);
        acc = wmma_bf16(af, bf_, acc);
      }
#pragma unroll
      for (int r = 0; r < 8; ++r) {
        size_t row = (size_t)(b * TT + c0 + tm + r + 8 * hi_);
        out[row * VD + h * DV + tn + n_] = acc[r];
      }
    }
    // ---- state update: S = lamC*S + Vs @ Kt^T  (M=v, N=d, K=s) ----
    for (int tt = wave; tt < 64; tt += 8) {
      int tm = (tt >> 3) * 16;   // v block
      int tn = (tt & 7) * 16;    // d block
      v8f acc;
#pragma unroll
      for (int r = 0; r < 8; ++r)
        acc[r] = lamC * Sf[(tm + r + 8 * hi_) * DK + tn + n_];
#pragma unroll
      for (int kk = 0; kk < CHUNK; kk += 32) {
        v16bf af  = frag_a(Vs + tm * CHUNK + kk, CHUNK, lane);
        v16bf bf_ = frag_bt(Kt + tn * CHUNK + kk, CHUNK, lane);
        acc = wmma_bf16(af, bf_, acc);
      }
#pragma unroll
      for (int r = 0; r < 8; ++r)
        Sf[(tm + r + 8 * hi_) * DK + tn + n_] = acc[r];
    }
    __syncthreads();
  }
}

// --------------------------- output gate + convert -------------------------
__global__ void gate_bf16(const float* __restrict__ attn,
                          const float* __restrict__ gpre,
                          __bf16* __restrict__ out, size_t n) {
  size_t i = (size_t)blockIdx.x * blockDim.x + threadIdx.x;
  if (i < n) out[i] = (__bf16)(attn[i] * sigmoidf_(gpre[i]));
}

// ---------------------------------------------------------------------------
extern "C" void kernel_launch(void* const* d_in, const int* in_sizes, int n_in,
                              void* d_out, int out_size, void* d_ws, size_t ws_size,
                              hipStream_t stream) {
  (void)in_sizes; (void)n_in; (void)out_size; (void)ws_size;
  const float* x      = (const float*)d_in[0];
  const float* Wq     = (const float*)d_in[1];
  const float* Wk     = (const float*)d_in[2];
  const float* Wv     = (const float*)d_in[3];
  const float* Wo     = (const float*)d_in[4];
  const float* Wg     = (const float*)d_in[5];
  const float* qconvw = (const float*)d_in[6];
  const float* qconvb = (const float*)d_in[7];
  const float* kconvw = (const float*)d_in[8];
  const float* kconvb = (const float*)d_in[9];
  const float* vconvw = (const float*)d_in[10];
  const float* vconvb = (const float*)d_in[11];
  const float* beta   = (const float*)d_in[12];
  const float* A_log  = (const float*)d_in[13];
  float* outp = (float*)d_out;

  const size_t NX = (size_t)BB * TT * DIMM;      // 8,388,608 activations
  const size_t NW = (size_t)DIMM * DIMM;         // 4,194,304 weight elems

  // ---- workspace carve ----
  char* ws = (char*)d_ws;
  __bf16* x_bf  = (__bf16*)ws;                    ws += NX * 2;
  __bf16* Wq_bf = (__bf16*)ws;                    ws += NW * 2;
  __bf16* Wk_bf = (__bf16*)ws;                    ws += NW * 2;
  __bf16* Wv_bf = (__bf16*)ws;                    ws += NW * 2;
  __bf16* Wg_bf = (__bf16*)ws;                    ws += NW * 2;
  __bf16* Wo_bf = (__bf16*)ws;                    ws += NW * 2;
  float* q_pre  = (float*)ws;                     ws += NX * 4;  // reused as attn
  float* k_pre  = (float*)ws;                     ws += NX * 4;
  float* v_pre  = (float*)ws;                     ws += NX * 4;
  float* g_pre  = (float*)ws;                     ws += NX * 4;
  __bf16* q_bf  = (__bf16*)ws;                    ws += NX * 2;
  __bf16* k_bf  = (__bf16*)ws;                    ws += NX * 2;
  __bf16* v_bf  = (__bf16*)ws;                    ws += NX * 2;
  float*  attn  = q_pre;      // q_pre dead after conv -> attn f32
  __bf16* y_bf  = x_bf;       // x_bf dead after projections -> gated bf16

  // ---- 1. converts ----
  cvt_f32_bf16<<<(int)(NX / 256), 256, 0, stream>>>(x, x_bf, NX);
  cvt_f32_bf16<<<(int)(NW / 256), 256, 0, stream>>>(Wq, Wq_bf, NW);
  cvt_f32_bf16<<<(int)(NW / 256), 256, 0, stream>>>(Wk, Wk_bf, NW);
  cvt_f32_bf16<<<(int)(NW / 256), 256, 0, stream>>>(Wv, Wv_bf, NW);
  cvt_f32_bf16<<<(int)(NW / 256), 256, 0, stream>>>(Wg, Wg_bf, NW);
  cvt_f32_bf16<<<(int)(NW / 256), 256, 0, stream>>>(Wo, Wo_bf, NW);

  // ---- 2. projections (4096x2048x2048 each) ----
  dim3 ggrid(DIMM / 128, (BB * TT) / 128);
  gemm_bf16_wmma<<<ggrid, 256, 0, stream>>>(x_bf, Wq_bf, q_pre, BB * TT, DIMM, DIMM);
  gemm_bf16_wmma<<<ggrid, 256, 0, stream>>>(x_bf, Wk_bf, k_pre, BB * TT, DIMM, DIMM);
  gemm_bf16_wmma<<<ggrid, 256, 0, stream>>>(x_bf, Wv_bf, v_pre, BB * TT, DIMM, DIMM);
  gemm_bf16_wmma<<<ggrid, 256, 0, stream>>>(x_bf, Wg_bf, g_pre, BB * TT, DIMM, DIMM);

  // ---- 3. causal conv + activation ----
  conv_causal<<<(int)(NX / 256), 256, 0, stream>>>(q_pre, qconvw, qconvb, q_bf, 1);
  conv_causal<<<(int)(NX / 256), 256, 0, stream>>>(k_pre, kconvw, kconvb, k_bf, 1);
  conv_causal<<<(int)(NX / 256), 256, 0, stream>>>(v_pre, vconvw, vconvb, v_bf, 0);

  // ---- 4. k L2 normalize per (b,t,h) ----
  knorm<<<(BB * TT * HH) / 8, 256, 0, stream>>>(k_bf);

  // ---- 5. chunked recurrence (one block per (b,h)); 168KB dynamic LDS ----
  const int lds_bytes = DV * DK * 4 + DV * DK * 2 + 4 * (CHUNK * DK * 2) + CHUNK * CHUNK * 2;
  (void)hipFuncSetAttribute((const void*)chunk_rec,
                            hipFuncAttributeMaxDynamicSharedMemorySize, lds_bytes);
  chunk_rec<<<BB * HH, 256, lds_bytes, stream>>>(q_bf, k_bf, v_bf, beta, A_log, attn);

  // ---- 6. output gate, convert to bf16 ----
  gate_bf16<<<(int)(NX / 256), 256, 0, stream>>>(attn, g_pre, y_bf, NX);

  // ---- 7. final projection out = y @ Wo ----
  gemm_bf16_wmma<<<ggrid, 256, 0, stream>>>(y_bf, Wo_bf, outp, BB * TT, DIMM, DIMM);
}
